// Model_44152263803522
// MI455X (gfx1250) — compile-verified, matrix operand order checked by reading, not run on hardware
//
#include <hip/hip_runtime.h>
#include <hip/hip_bf16.h>

typedef __attribute__((ext_vector_type(16))) __bf16 v16bf;
typedef __attribute__((ext_vector_type(8)))  float  v8f;

#define WAVE 32
static __device__ __forceinline__ float atomic_add_f32(float* p, float v) {
    return __hip_atomic_fetch_add(p, v, __ATOMIC_RELAXED, __HIP_MEMORY_SCOPE_AGENT);
}

// ---------------------------------------------------------------- degree init
__global__ void k_deg_init(float* deg, int N) {
    int i = blockIdx.x * blockDim.x + threadIdx.x;
    if (i < N) deg[i] = 1.0f;                      // self-loop contributes 1
}

__global__ void k_deg_scatter(float* deg, const int* __restrict__ dst, int E) {
    int e = blockIdx.x * blockDim.x + threadIdx.x;
    if (e < E) atomic_add_f32(&deg[dst[e]], 1.0f);
}

__global__ void k_dinv(float* deg, int N) {        // in place: deg -> d^{-1/2}
    int i = blockIdx.x * blockDim.x + threadIdx.x;
    if (i < N) deg[i] = rsqrtf(fmaxf(deg[i], 1.0f));
}

__global__ void k_norm(float* __restrict__ norm, const float* __restrict__ dinv,
                       const int* __restrict__ src, const int* __restrict__ dst, int E) {
    int e = blockIdx.x * blockDim.x + threadIdx.x;
    if (e < E) norm[e] = dinv[src[e]] * dinv[dst[e]];
}

// ---------------------------------------------------------------- W transpose
// Wt[n*K + k] = W[k*Ncols + n]   (fp32)
__global__ void k_transpose(const float* __restrict__ W, float* __restrict__ Wt,
                            int K, int Ncols) {
    int i = blockIdx.x * blockDim.x + threadIdx.x;
    if (i >= K * Ncols) return;
    int k = i / Ncols, n = i - k * Ncols;
    Wt[(size_t)n * K + k] = W[i];
}

// ---------------------------------------------------------------- WMMA GEMM
// C[M x 128] = A[M x K] @ B[K x 128], B supplied transposed (BT[128 x K]).
// Split-precision: fp32 -> bf16 hi + bf16 lo residual in registers; 3 WMMAs
// per 32-wide k-step recover ~16 mantissa bits.
// Fragment K-map (16-bit A, wave32): element e of v16bf on lane L holds
//   K = kbase + sel*8 + e          for e in [0,8)   (sel = L>>4)
//   K = kbase + 16 + sel*8 + (e-8) for e in [8,16)
template <bool RELU>
__device__ __forceinline__ void load_split_frag(const float* __restrict__ rowptr,
                                                int k, v16bf& hi, v16bf& lo) {
    float buf[16];
    float4 p0 = *(const float4*)(rowptr + k);
    float4 p1 = *(const float4*)(rowptr + k + 4);
    float4 p2 = *(const float4*)(rowptr + k + 16);
    float4 p3 = *(const float4*)(rowptr + k + 20);
    buf[0]=p0.x; buf[1]=p0.y; buf[2]=p0.z; buf[3]=p0.w;
    buf[4]=p1.x; buf[5]=p1.y; buf[6]=p1.z; buf[7]=p1.w;
    buf[8]=p2.x; buf[9]=p2.y; buf[10]=p2.z; buf[11]=p2.w;
    buf[12]=p3.x; buf[13]=p3.y; buf[14]=p3.z; buf[15]=p3.w;
#pragma unroll
    for (int e = 0; e < 16; ++e) {
        float v = buf[e];
        if (RELU) v = fmaxf(v, 0.0f);
        __bf16 h = (__bf16)v;
        hi[e] = h;
        lo[e] = (__bf16)(v - (float)h);
    }
}

template <bool RELU_A>
__global__ __launch_bounds__(256) void k_gemm(const float* __restrict__ A,
                                              const float* __restrict__ BT,
                                              float* __restrict__ C,
                                              int K) {   // N fixed = 128
    const int wave = threadIdx.x >> 5;
    const int lane = threadIdx.x & 31;
    const int m0 = blockIdx.x * 16;
    const int n0 = wave * 16;
    const int sel = lane >> 4;

    const float* arow = A  + (size_t)(m0 + (lane & 15)) * K + sel * 8;
    const float* brow = BT + (size_t)(n0 + (lane & 15)) * K + sel * 8;

    v8f acc = {};
    for (int k = 0; k < K; k += 32) {
        v16bf ah, al, bh, bl;
        load_split_frag<RELU_A>(arow, k, ah, al);
        load_split_frag<false >(brow, k, bh, bl);
        acc = __builtin_amdgcn_wmma_f32_16x16x32_bf16(false, ah, false, bh,
                                                      (short)0, acc, false, false);
        acc = __builtin_amdgcn_wmma_f32_16x16x32_bf16(false, ah, false, bl,
                                                      (short)0, acc, false, false);
        acc = __builtin_amdgcn_wmma_f32_16x16x32_bf16(false, al, false, bh,
                                                      (short)0, acc, false, false);
    }
    // C/D layout: VGPR r -> M = m0 + 8*sel + r, N = n0 + (lane&15)
    const int n = n0 + (lane & 15);
    const int mbase = m0 + sel * 8;
#pragma unroll
    for (int r = 0; r < 8; ++r)
        C[(size_t)(mbase + r) * 128 + n] = acc[r];
}

// ---------------------------------------------------------------- propagation
// znew = (1-a) * selfnorm * z  +  a * h       (self-loop + teleport)
__global__ void k_prop_init(float* __restrict__ znew, const float* __restrict__ z,
                            const float* __restrict__ h, const float* __restrict__ dinv,
                            long total) {
    long i = (long)blockIdx.x * blockDim.x + threadIdx.x;
    if (i >= total) return;
    float sn = dinv[i >> 7];
    sn = sn * sn;
    znew[i] = 0.9f * sn * z[i] + 0.1f * h[i];
}

// znew[dst] += (1-a) * norm[e] * z[src]   — one wave per edge, float4 per lane
__global__ __launch_bounds__(256) void k_prop_edges(float* __restrict__ znew,
                                                    const float* __restrict__ z,
                                                    const int* __restrict__ src,
                                                    const int* __restrict__ dst,
                                                    const float* __restrict__ norm,
                                                    int E) {
    int gw = (int)(((long)blockIdx.x * blockDim.x + threadIdx.x) >> 5);
    if (gw >= E) return;
    int lane = threadIdx.x & 31;
    int s = src[gw], d = dst[gw];
    float w = 0.9f * norm[gw];
    float4 v = ((const float4*)(z + (size_t)s * 128))[lane];
    float* p = znew + (size_t)d * 128 + lane * 4;
    atomic_add_f32(p + 0, w * v.x);
    atomic_add_f32(p + 1, w * v.y);
    atomic_add_f32(p + 2, w * v.z);
    atomic_add_f32(p + 3, w * v.w);
}

// ---------------------------------------------------------------- decoder
__global__ __launch_bounds__(256) void k_decode(const float* __restrict__ z,
                                                const int* __restrict__ pos,
                                                const int* __restrict__ neg,
                                                float* __restrict__ out, int ELP) {
    int gw = (int)(((long)blockIdx.x * blockDim.x + threadIdx.x) >> 5);
    if (gw >= 2 * ELP) return;
    int lane = threadIdx.x & 31;
    int a, b;
    if (gw < ELP) { a = pos[gw];        b = pos[ELP + gw]; }
    else          { int e = gw - ELP; a = neg[e]; b = neg[ELP + e]; }
    float4 va = ((const float4*)(z + (size_t)a * 128))[lane];
    float4 vb = ((const float4*)(z + (size_t)b * 128))[lane];
    float p = va.x * vb.x + va.y * vb.y + va.z * vb.z + va.w * vb.w;
#pragma unroll
    for (int off = 16; off; off >>= 1) p += __shfl_down(p, off, 32);
    if (lane == 0) out[gw] = p;
}

// ---------------------------------------------------------------- launcher
static inline size_t align256(size_t x) { return (x + 255) & ~(size_t)255; }

extern "C" void kernel_launch(void* const* d_in, const int* in_sizes, int n_in,
                              void* d_out, int out_size, void* d_ws, size_t ws_size,
                              hipStream_t stream) {
    const float* x   = (const float*)d_in[0];
    const int*   ei  = (const int*)  d_in[1];
    const int*   pos = (const int*)  d_in[2];
    const int*   neg = (const int*)  d_in[3];
    const float* W1  = (const float*)d_in[4];
    const float* W2  = (const float*)d_in[5];
    float* out = (float*)d_out;

    const int H   = 128;
    const int Fin = 256;
    const int N   = in_sizes[0] / Fin;    // 100000
    const int E   = in_sizes[1] / 2;      // 1.6M
    const int ELP = in_sizes[2] / 2;      // 400K

    // -------- workspace carve-up
    char* w = (char*)d_ws;
    size_t off = 0;
    float* dinv = (float*)(w + off); off = align256(off + (size_t)N * 4);
    float* norm = (float*)(w + off); off = align256(off + (size_t)E * 4);
    float* Wt   = (float*)(w + off); off = align256(off + (size_t)Fin * H * 4);
    float* h    = (float*)(w + off); off = align256(off + (size_t)N * H * 4);
    float* zA   = (float*)(w + off); off = align256(off + (size_t)N * H * 4);
    float* zB   = (float*)(w + off); off = align256(off + (size_t)N * H * 4);
    (void)ws_size;

    const int* src = ei;
    const int* dst = ei + E;
    const long tot = (long)N * H;

    // -------- GCN normalization
    k_deg_init<<<(N + 255) / 256, 256, 0, stream>>>(dinv, N);
    k_deg_scatter<<<(E + 255) / 256, 256, 0, stream>>>(dinv, dst, E);
    k_dinv<<<(N + 255) / 256, 256, 0, stream>>>(dinv, N);
    k_norm<<<(E + 255) / 256, 256, 0, stream>>>(norm, dinv, src, dst, E);

    // -------- layer 1: h = x @ W1, then K=10 APPNP hops
    k_transpose<<<(Fin * H + 255) / 256, 256, 0, stream>>>(W1, Wt, Fin, H);
    k_gemm<false><<<N / 16, 256, 0, stream>>>(x, Wt, h, Fin);

    const float* zin = h;
    for (int hop = 0; hop < 10; ++hop) {
        float* zout = (hop & 1) ? zB : zA;
        k_prop_init<<<(int)((tot + 255) / 256), 256, 0, stream>>>(zout, zin, h, dinv, tot);
        k_prop_edges<<<(E + 7) / 8, 256, 0, stream>>>(zout, zin, src, dst, norm, E);
        zin = zout;
    }

    // -------- layer 2: h2 = relu(z) @ W2 (ReLU fused into GEMM A-loads)
    k_transpose<<<(H * H + 255) / 256, 256, 0, stream>>>(W2, Wt, H, H);
    k_gemm<true><<<N / 16, 256, 0, stream>>>(zin, Wt, h, H);

    zin = h;
    for (int hop = 0; hop < 10; ++hop) {
        float* zout = (hop & 1) ? zB : zA;
        k_prop_init<<<(int)((tot + 255) / 256), 256, 0, stream>>>(zout, zin, h, dinv, tot);
        k_prop_edges<<<(E + 7) / 8, 256, 0, stream>>>(zout, zin, src, dst, norm, E);
        zin = zout;
    }

    // -------- decode: edge dot products
    k_decode<<<(2 * ELP + 7) / 8, 256, 0, stream>>>(zin, pos, neg, out, ELP);
}
